// HyperedgeGuidedNodeAttention_28174985462031
// MI455X (gfx1250) — matrix operand order repaired and checked by reading, not compiled, not run on hardware
//
#include <hip/hip_runtime.h>

// Problem dims
constexpr int BB = 8;
constexpr int NN = 4096;
constexpr int EE = 1024;
constexpr int CC = 512;

typedef __attribute__((ext_vector_type(16))) __bf16 v16bf;
typedef __attribute__((ext_vector_type(8)))  float  v8f;
typedef __attribute__((ext_vector_type(4)))  unsigned int u32x4;
typedef __attribute__((ext_vector_type(8)))  int i32x8;
typedef __attribute__((ext_vector_type(4)))  int i32x4;

union Frag {
    v16bf v;
    uint4 q[2];
};

// ---------------------------------------------------------------------------
// Projection GEMM core: one wave computes a 16x16 tile of  Out = A @ W^T + b
// A: (M,K) fp32 row-major, W: (Nout,K) fp32 row-major.
// WMMA bf16 fragments:
//   A-frag (16x32): lane l -> row = l&15, K chunks at kbA={0|8} and kbA+16
//   B-frag (32x16): lane l -> col = l&15, K chunks at kbB={0|8} and kbB+16
// ---------------------------------------------------------------------------
__device__ inline v8f proj_tile(const float* __restrict__ A,
                                const float* __restrict__ W,
                                const float* __restrict__ bias,
                                int m0, int n0, int K, int lane) {
    const int c15 = lane & 15;
    const int hf  = lane >> 4;
    const int kbA = hf * 8;
    const int kbB = hf * 8;

    v8f acc;
    const float bv = bias[n0 + c15];
#pragma unroll
    for (int g = 0; g < 8; ++g) acc[g] = bv;

#pragma unroll 4
    for (int k0 = 0; k0 < K; k0 += 32) {
        Frag fa, fb;
        const float* ap = A + (size_t)(m0 + c15) * K + k0 + kbA;
#pragma unroll
        for (int i = 0; i < 8; ++i) fa.v[i] = (__bf16)ap[i];
#pragma unroll
        for (int i = 0; i < 8; ++i) fa.v[8 + i] = (__bf16)ap[16 + i];

        const float* wp = W + (size_t)(n0 + c15) * K + k0 + kbB;
#pragma unroll
        for (int i = 0; i < 8; ++i) fb.v[i] = (__bf16)wp[i];
#pragma unroll
        for (int i = 0; i < 8; ++i) fb.v[8 + i] = (__bf16)wp[16 + i];

        acc = __builtin_amdgcn_wmma_f32_16x16x32_bf16(
            false, fa.v, false, fb.v, (short)0, acc, false, false);
    }
    return acc;
}

// Q projection: Out (B*N, C) bf16 row-major
__global__ __launch_bounds__(256) void proj_q_kernel(const float* __restrict__ A,
                                                     const float* __restrict__ W,
                                                     const float* __restrict__ bias,
                                                     __bf16* __restrict__ Out) {
    const int lane = threadIdx.x & 31;
    const int wid  = (blockIdx.x * blockDim.x + threadIdx.x) >> 5;
    const int tiles_n = CC / 16;                 // 32
    const int tiles_m = (BB * NN) / 16;          // 2048
    if (wid >= tiles_m * tiles_n) return;
    const int m0 = (wid / tiles_n) * 16;
    const int n0 = (wid % tiles_n) * 16;

    v8f acc = proj_tile(A, W, bias, m0, n0, CC, lane);

    const int c15 = lane & 15;
    const int hf  = lane >> 4;
#pragma unroll
    for (int g = 0; g < 8; ++g) {
        const int r = m0 + g + 8 * hf;
        Out[(size_t)r * CC + n0 + c15] = (__bf16)acc[g];
    }
}

// KV projection: K -> (B,E,C) bf16 row-major; V -> (B,C,E) bf16 (transposed)
__global__ __launch_bounds__(256) void proj_kv_kernel(const float* __restrict__ A,
                                                      const float* __restrict__ W,
                                                      const float* __restrict__ bias,
                                                      __bf16* __restrict__ Kout,
                                                      __bf16* __restrict__ Vtout) {
    const int lane = threadIdx.x & 31;
    const int wid  = (blockIdx.x * blockDim.x + threadIdx.x) >> 5;
    const int tiles_n = (2 * CC) / 16;           // 64
    const int tiles_m = (BB * EE) / 16;          // 512
    if (wid >= tiles_m * tiles_n) return;
    const int m0 = (wid / tiles_n) * 16;
    const int n0 = (wid % tiles_n) * 16;

    v8f acc = proj_tile(A, W, bias, m0, n0, CC, lane);

    const int c15 = lane & 15;
    const int hf  = lane >> 4;
#pragma unroll
    for (int g = 0; g < 8; ++g) {
        const int r = m0 + g + 8 * hf;           // global row in (B*E)
        const int bi = r / EE;
        const int e  = r % EE;
        const int n  = n0 + c15;
        if (n0 < CC) {
            Kout[((size_t)bi * EE + e) * CC + n] = (__bf16)acc[g];
        } else {
            Vtout[((size_t)bi * CC + (n - CC)) * EE + e] = (__bf16)acc[g];
        }
    }
}

// ---------------------------------------------------------------------------
// Tensor Data Mover: 2-D tile load into LDS with row padding (ISA ch.8 D#).
//   lds_byte : LDS byte offset of destination
//   gaddr    : global byte address of tile start
//   g1d0     : group1 dword0 (data_size / pad config)
//   td0,td1  : tensor_dim0/1 (elements);  tile0,tile1 : tile_dim0/1
//   stride0  : tensor_dim0_stride (elements)
// ---------------------------------------------------------------------------
__device__ inline unsigned ldsOff(const void* p) {
    // Generic (flat) LDS pointers carry the LDS byte offset in addr[31:0].
    return (unsigned)(size_t)p;
}

__device__ inline void tdm_load_2d(unsigned lds_byte, const void* gaddr,
                                   unsigned g1d0, unsigned td0, unsigned td1,
                                   unsigned tile0, unsigned tile1,
                                   unsigned long long stride0) {
    const unsigned long long ga = (unsigned long long)(size_t)gaddr;
    u32x4 g0;
    g0.x = 1u;                                     // count=1, user mode
    g0.y = lds_byte;                               // lds_addr
    g0.z = (unsigned)(ga & 0xFFFFFFFFu);           // global_addr[31:0]
    g0.w = (unsigned)(ga >> 32) | (2u << 30);      // global_addr[56:32] | type=2
    i32x8 g1;
    g1[0] = (int)g1d0;
    g1[1] = (int)((td0 & 0xFFFFu) << 16);                      // tensor_dim0[15:0]
    g1[2] = (int)((td0 >> 16) | ((td1 & 0xFFFFu) << 16));      // td0 hi | td1 lo
    g1[3] = (int)((td1 >> 16) | (tile0 << 16));                // td1 hi | tile_dim0
    g1[4] = (int)(tile1 & 0xFFFFu);                            // tile_dim1, tile_dim2=0
    g1[5] = (int)(unsigned)(stride0 & 0xFFFFFFFFull);          // dim0 stride lo
    g1[6] = (int)(unsigned)(stride0 >> 32);                    // dim0 stride hi
    g1[7] = 0;
    const i32x4 z4 = {0, 0, 0, 0};
    const i32x8 z8 = {0, 0, 0, 0, 0, 0, 0, 0};
    // clang-23 / amdgpu-toolchain 6-arg form:
    //   (g0: v4u32, g1: v8i32, g2: v4i32, g3: v4i32, ext: v8i32, cpol: i32)
    __builtin_amdgcn_tensor_load_to_lds(g0, g1, z4, z4, z8, 0);
}

// data_size=2B (1<<16) | pad_enable (1<<20) | pad_interval | pad_amount(4dw -> 3)
constexpr unsigned G1_K = (1u << 16) | (1u << 20) | (7u << 22) | (3u << 25); // pad every 256dw
constexpr unsigned G1_V = (1u << 16) | (1u << 20) | (3u << 22) | (3u << 25); // pad every 16dw

// ---------------------------------------------------------------------------
// Flash attention: 4 waves/WG, each wave owns 16 Q rows, full C=512 output.
// E streamed in chunks of 32 via double-buffered TDM DMA. Online softmax.
// ---------------------------------------------------------------------------
constexpr int ECH   = 32;
constexpr int KS_LD = CC + 8;    // 520 bf16 = 1040B rows: 16B aligned, bank-skewed
constexpr int VT_LD = ECH + 8;   // 40 bf16 = 80B rows:   16B aligned, bank-skewed
constexpr int NCH   = EE / ECH;  // 32 chunks

__global__ __launch_bounds__(128) void flash_kernel(const __bf16* __restrict__ Qb,
                                                    const __bf16* __restrict__ Kb,
                                                    const __bf16* __restrict__ Vtb,
                                                    float* __restrict__ Out) {
    __shared__ __align__(16) __bf16 Ks[2][ECH][KS_LD];   // 2 x 32.5 KB
    __shared__ __align__(16) __bf16 Vts[2][CC][VT_LD];   // 2 x 40 KB
    __shared__ __align__(16) __bf16 Ps[4][16][VT_LD];    // 5 KB

    const int tid  = threadIdx.x;
    const int lane = tid & 31;
    const int wave = tid >> 5;
    const int c15  = lane & 15;
    const int hf   = lane >> 4;
    const int kbA  = hf * 8;
    const int kbB  = hf * 8;

    const int blocksPerB = NN / 64;              // 64
    const int b  = blockIdx.x / blocksPerB;
    const int m0 = (blockIdx.x % blocksPerB) * 64 + wave * 16;

    const __bf16* kbase = Kb + (size_t)b * EE * CC;
    const __bf16* vbase = Vtb + (size_t)b * CC * EE;

    // Resident Q A-fragments: 16 frags x 32 K each = full C=512
    Frag qf[16];
    {
        const __bf16* qrow = Qb + ((size_t)b * NN + m0 + c15) * CC;
#pragma unroll
        for (int f = 0; f < 16; ++f) {
            qf[f].q[0] = *reinterpret_cast<const uint4*>(qrow + f * 32 + kbA);
            qf[f].q[1] = *reinterpret_cast<const uint4*>(qrow + f * 32 + kbA + 16);
        }
    }

    v8f acc[32];
#pragma unroll
    for (int t = 0; t < 32; ++t)
#pragma unroll
        for (int g = 0; g < 8; ++g) acc[t][g] = 0.f;

    float m_row[8], l_row[8];
#pragma unroll
    for (int g = 0; g < 8; ++g) { m_row[g] = -1e30f; l_row[g] = 0.f; }

    // scale * log2(e): softmax done in exp2 domain
    const float sc = 0.044194173824159216f * 1.4426950408889634f;

    // Prologue: DMA chunk 0 into buffer 0 (one wave drives the TDM)
    if (wave == 0) {
        tdm_load_2d(ldsOff(&Ks[0][0][0]),  kbase,     G1_K, CC,  ECH, CC,  ECH, CC);
        tdm_load_2d(ldsOff(&Vts[0][0][0]), vbase,     G1_V, ECH, CC,  ECH, CC,  EE);
    }

    for (int c = 0; c < NCH; ++c) {
        const int cur = c & 1;
        if (wave == 0) {
            if (c + 1 < NCH) {
                const int e1 = (c + 1) * ECH;
                tdm_load_2d(ldsOff(&Ks[cur ^ 1][0][0]),  kbase + (size_t)e1 * CC,
                            G1_K, CC,  ECH, CC,  ECH, CC);
                tdm_load_2d(ldsOff(&Vts[cur ^ 1][0][0]), vbase + e1,
                            G1_V, ECH, CC,  ECH, CC,  EE);
                __builtin_amdgcn_s_wait_tensorcnt(2);  // two newest still in flight
            } else {
                __builtin_amdgcn_s_wait_tensorcnt(0);  // final chunk landed
            }
        }
        __syncthreads();   // chunk `c` visible to all waves

        // ---- S = Q K^T for 16 rows x 32 e-cols ----
        v8f s0, s1;
#pragma unroll
        for (int g = 0; g < 8; ++g) { s0[g] = 0.f; s1[g] = 0.f; }
#pragma unroll
        for (int f = 0; f < 16; ++f) {
            Frag k0, k1;
            k0.q[0] = *reinterpret_cast<const uint4*>(&Ks[cur][c15][f * 32 + kbB]);
            k0.q[1] = *reinterpret_cast<const uint4*>(&Ks[cur][c15][f * 32 + kbB + 16]);
            k1.q[0] = *reinterpret_cast<const uint4*>(&Ks[cur][16 + c15][f * 32 + kbB]);
            k1.q[1] = *reinterpret_cast<const uint4*>(&Ks[cur][16 + c15][f * 32 + kbB + 16]);
            s0 = __builtin_amdgcn_wmma_f32_16x16x32_bf16(
                false, qf[f].v, false, k0.v, (short)0, s0, false, false);
            s1 = __builtin_amdgcn_wmma_f32_16x16x32_bf16(
                false, qf[f].v, false, k1.v, (short)0, s1, false, false);
        }
#pragma unroll
        for (int g = 0; g < 8; ++g) { s0[g] *= sc; s1[g] *= sc; }

        // ---- online softmax (row = g + 8*hf, cols across 16-lane halves) ----
        float alpha[8];
#pragma unroll
        for (int g = 0; g < 8; ++g) {
            float mx = fmaxf(s0[g], s1[g]);
#pragma unroll
            for (int off = 1; off < 16; off <<= 1)
                mx = fmaxf(mx, __shfl_xor(mx, off, 32));
            const float mn = fmaxf(m_row[g], mx);
            alpha[g] = exp2f(m_row[g] - mn);
            m_row[g] = mn;
        }
#pragma unroll
        for (int g = 0; g < 8; ++g) {
            const float p0 = exp2f(s0[g] - m_row[g]);
            const float p1 = exp2f(s1[g] - m_row[g]);
            const int rD = g + 8 * hf;
            Ps[wave][rD][c15]      = (__bf16)p0;
            Ps[wave][rD][16 + c15] = (__bf16)p1;
            float rs = p0 + p1;
#pragma unroll
            for (int off = 1; off < 16; off <<= 1)
                rs += __shfl_xor(rs, off, 32);
            l_row[g] = l_row[g] * alpha[g] + rs;
        }
#pragma unroll
        for (int t = 0; t < 32; ++t)
#pragma unroll
            for (int g = 0; g < 8; ++g) acc[t][g] *= alpha[g];

        // ---- reload P as bf16 A-fragment (same-wave LDS round trip) ----
        Frag pf;
        pf.q[0] = *reinterpret_cast<const uint4*>(&Ps[wave][c15][kbA]);
        pf.q[1] = *reinterpret_cast<const uint4*>(&Ps[wave][c15][kbA + 16]);

        // ---- acc += P @ V  (B-frags from transposed V in LDS) ----
#pragma unroll
        for (int t = 0; t < 32; ++t) {
            Frag vb;
            vb.q[0] = *reinterpret_cast<const uint4*>(&Vts[cur][t * 16 + c15][kbB]);
            vb.q[1] = *reinterpret_cast<const uint4*>(&Vts[cur][t * 16 + c15][kbB + 16]);
            acc[t] = __builtin_amdgcn_wmma_f32_16x16x32_bf16(
                false, pf.v, false, vb.v, (short)0, acc[t], false, false);
        }
        __syncthreads();   // buffer `cur` free for the DMA issued next iteration
    }

    // ---- epilogue: normalize and store fp32 ----
#pragma unroll
    for (int t = 0; t < 32; ++t) {
#pragma unroll
        for (int g = 0; g < 8; ++g) {
            const int rD = g + 8 * hf;
            Out[((size_t)b * NN + m0 + rD) * CC + t * 16 + c15] = acc[t][g] / l_row[g];
        }
    }
}

// ---------------------------------------------------------------------------
extern "C" void kernel_launch(void* const* d_in, const int* in_sizes, int n_in,
                              void* d_out, int out_size, void* d_ws, size_t ws_size,
                              hipStream_t stream) {
    const float* node  = (const float*)d_in[0];
    const float* hyper = (const float*)d_in[1];
    const float* Wq    = (const float*)d_in[2];
    const float* bq    = (const float*)d_in[3];
    const float* Wkv   = (const float*)d_in[4];
    const float* bkv   = (const float*)d_in[5];
    float* out = (float*)d_out;

    // workspace: Qb (32MB) | Kb (8MB) | Vtb (8MB), all bf16
    __bf16* Qb  = (__bf16*)d_ws;
    __bf16* Kb  = Qb + (size_t)BB * NN * CC;
    __bf16* Vtb = Kb + (size_t)BB * EE * CC;

    // Q projection: 2048*32 = 65536 tiles, 8 waves/block -> 8192 blocks
    proj_q_kernel<<<8192, 256, 0, stream>>>(node, Wq, bq, Qb);
    // KV projection: 512*64 = 32768 tiles -> 4096 blocks
    proj_kv_kernel<<<4096, 256, 0, stream>>>(hyper, Wkv, bkv, Kb, Vtb);
    // Flash attention: 8 batches * 64 row-blocks = 512 WGs of 128 threads
    flash_kernel<<<BB * (NN / 64), 128, 0, stream>>>(Qb, Kb, Vtb, out);
}
